// SwinTransformerBlock3D_23441931502132
// MI455X (gfx1250) — matrix operand order repaired and checked by reading, not compiled, not run on hardware
//
#include <hip/hip_runtime.h>
#include <math.h>

// ---------------------------------------------------------------------------
// Swin-3D block for gfx1250 (MI455X). bf16 WMMA (v_wmma_f32_16x16x32_bf16)
// for all GEMMs, TDM (tensor_load_to_lds) for A-tile staging, f32 LN/softmax.
// ---------------------------------------------------------------------------

typedef unsigned short u16;
typedef unsigned int   u32;
typedef unsigned long long u64;
typedef __attribute__((ext_vector_type(16))) __bf16 v16bf;
typedef __attribute__((ext_vector_type(8)))  float  v8f;
typedef __attribute__((ext_vector_type(4)))  unsigned int v4u;
typedef __attribute__((ext_vector_type(4)))  int v4i;
typedef __attribute__((ext_vector_type(8)))  int v8i;

// problem constants
#define BB    2
#define CC    192
#define ZD    10
#define HD_   48
#define WD    48
#define WZ    5
#define WH    8
#define WWIN  8
#define SZ    2
#define SH    4
#define SW    4
#define NWIN  320            // WZ*WH*WWIN
#define NZW   2              // ZD/WZ
#define NHW   6              // HD_/WH
#define NWW   6              // WD/WWIN
#define NWTOT 144            // BB*NZW*NHW*NWW
#define NHEAD 6
#define HDIM  32
#define C3    576
#define HID   768
#define MTOK  46080          // BB*ZD*HD_*WD
#define SPAT  23040          // ZD*HD_*WD

static __device__ __forceinline__ u16 f2bf(float f) {
  u32 x = __float_as_uint(f);
  x += 0x7fffu + ((x >> 16) & 1u);           // round-to-nearest-even
  return (u16)(x >> 16);
}

union Frag16 { uint4 q[2]; v16bf v; };

// Build a v16bf A/B fragment from two 16-byte chunks (ISA 7.12.2 layouts).
static __device__ __forceinline__ v16bf load_frag2(const u16* lo, const u16* hi) {
  Frag16 f;
  f.q[0] = *reinterpret_cast<const uint4*>(lo);
  f.q[1] = *reinterpret_cast<const uint4*>(hi);
  return f.v;
}

static __device__ __forceinline__ v8f wmma_bf16(v16bf a, v16bf b, v8f c) {
  return __builtin_amdgcn_wmma_f32_16x16x32_bf16(false, a, false, b, (short)0, c,
                                                 false, false);
}

// ---------------------------------------------------------------------------
// TDM: 2D tile load Global -> LDS.  tile_x elements (2B each) per row,
// tile_y rows, row stride = stride_elems. No OOB (tensor dims == tile dims).
// D# layout per CDNA5 ISA ch.8: group0 {count,lds,global,type=2},
// group1 {data_size=1, dims/strides}, groups 2/3 zero (2-D tensor).
// ---------------------------------------------------------------------------
static __device__ __forceinline__ void tdm_load_2d(u32 lds_addr, const void* gptr,
                                                   u32 tile_x, u32 tile_y,
                                                   u32 stride_elems) {
  union { u32 w[4]; v4u v; } g0;
  union { u64 q[4]; v8i v; } g1;
  u64 ga = (u64)gptr;
  g0.w[0] = 1u;                                            // count=1 (user D#)
  g0.w[1] = lds_addr;                                      // LDS byte address
  g0.w[2] = (u32)ga;                                       // global_addr[31:0]
  g0.w[3] = ((u32)(ga >> 32) & 0x1ffffffu) | (2u << 30);   // ga[56:32] | type=2

  u64 td0 = tile_x, td1 = tile_y, st0 = stride_elems;
  g1.q[0] = (1ull << 16) | (td0 << 48);                    // data_size=1 (2B); tensor_dim0 lo16
  g1.q[1] = (td0 >> 16) | (td1 << 16) | ((u64)tile_x << 48); // t_dim0 hi | t_dim1 | tile_dim0
  g1.q[2] = (u64)tile_y | ((st0 & 0xffffffffull) << 32);   // tile_dim1 | stride0 lo32
  g1.q[3] = (st0 >> 32);                                   // stride0 hi16
  v4i z4 = {};
#if defined(__clang_major__) && (__clang_major__ >= 23)
  v8i z8 = {};
  __builtin_amdgcn_tensor_load_to_lds(g0.v, g1.v, z4, z4, z8, 0);
#else
  __builtin_amdgcn_tensor_load_to_lds(g0.v, g1.v, z4, z4, 0);
#endif
}

// LDS byte offset of a generic pointer into shared memory (flat-aperture rule:
// low 32 bits of a generic LDS address are the LDS byte address).
static __device__ __forceinline__ u32 lds_off(const void* p) {
  return (u32)(unsigned long long)p;
}

// ---------------------------------------------------------------------------
// Kernel: f32 -> bf16 convert (weights)
// ---------------------------------------------------------------------------
__global__ void cvt_f32_bf16(const float* __restrict__ src, u16* __restrict__ dst,
                             int n) {
  int i = blockIdx.x * blockDim.x + threadIdx.x;
  if (i < n) dst[i] = f2bf(src[i]);
}

// ---------------------------------------------------------------------------
// Kernel: LayerNorm1 + cyclic shift + window partition -> bf16 (NWTOT,320,192)
// One thread per token; channel loop is coalesced across threads (x is CF).
// ---------------------------------------------------------------------------
__global__ void ln1_window_kernel(const float* __restrict__ x,
                                  const float* __restrict__ g,
                                  const float* __restrict__ b,
                                  u16* __restrict__ xwin) {
  int p = blockIdx.x * blockDim.x + threadIdx.x;
  if (p >= MTOK) return;
  int w = p % WD; int t = p / WD;
  int h = t % HD_; t /= HD_;
  int z = t % ZD;  int bb = t / ZD;

  const float* xp = x + (size_t)bb * CC * SPAT + (size_t)z * HD_ * WD + h * WD + w;
  float s = 0.f, ss = 0.f;
  for (int c = 0; c < CC; ++c) { float v = xp[(size_t)c * SPAT]; s += v; ss += v * v; }
  float mu = s * (1.0f / CC);
  float var = ss * (1.0f / CC) - mu * mu;
  float rs = rsqrtf(var + 1e-5f);

  int zr = z - SZ; if (zr < 0) zr += ZD;
  int hr = h - SH; if (hr < 0) hr += HD_;
  int wr = w - SW; if (wr < 0) wr += WD;
  int wz = zr / WZ, lz = zr % WZ;
  int wh = hr / WH, lh = hr % WH;
  int ww = wr / WWIN, lw = wr % WWIN;
  int bw = ((bb * NZW + wz) * NHW + wh) * NWW + ww;
  int n  = (lz * WH + lh) * WWIN + lw;

  u16* op = xwin + ((size_t)bw * NWIN + n) * CC;
  for (int c = 0; c < CC; ++c)
    op[c] = f2bf((xp[(size_t)c * SPAT] - mu) * rs * g[c] + b[c]);
}

// ---------------------------------------------------------------------------
// Generic bf16 GEMM:  out[m,n] = epi( sum_k A[m,k] * W[n,k] + bias[n] )
// 256 thr = 8 waves; block tile 64(M) x 128(N); wave tile 32x32:
//   2 A-frags x 2 B-frags -> 4 WMMAs per k-step (2x fragment reuse).
// Whole A block-tile (64 x Kd bf16) staged ONCE per block via TDM; the
// k-loop is then barrier-free (ds_load + global_load + wmma only).
// EPI 0: store bf16  | 1: GELU->bf16 | 2: proj residual + window-reverse ->f32
// EPI 3: fc2 residual + transpose (B,C,Z,H,W) -> f32
// ---------------------------------------------------------------------------
template <int EPI>
__global__ __launch_bounds__(256) void gemm_wmma(
    const u16* __restrict__ A, const u16* __restrict__ Wt,
    const float* __restrict__ bias, int Ncols, int Kd,
    void* __restrict__ out, const float* __restrict__ resid) {
  extern __shared__ u16 Atile[];   // [64][Kd]

  const int tid = threadIdx.x;
  const int wave = tid >> 5;
  const int lane = tid & 31;
  const int hi = lane >> 4;          // lane half
  const int ln = lane & 15;
  const int waveM = wave & 1;        // 2 M-groups of 32 rows
  const int waveN = wave >> 1;       // 4 N-groups of 32 cols
  const int m0 = blockIdx.x * 64;
  const int nb = blockIdx.y * 128 + waveN * 32;

  const bool t0ok = (nb + 16) <= Ncols;
  const bool t1ok = (nb + 32) <= Ncols;

  // ---- TDM: DMA the 64 x Kd A-tile into LDS (one descriptor, wave 0) ----
  if (wave == 0) {
    tdm_load_2d(lds_off(Atile), A + (size_t)m0 * Kd, (u32)Kd, 64u, (u32)Kd);
    __builtin_amdgcn_s_wait_tensorcnt(0);
  }
  __syncthreads();

  v8f acc00 = {}, acc01 = {}, acc10 = {}, acc11 = {};

  for (int k0 = 0; k0 < Kd; k0 += 32) {
    const u16* ar0 = &Atile[(size_t)(waveM * 32 + ln) * Kd + k0];
    const u16* ar1 = &Atile[(size_t)(waveM * 32 + 16 + ln) * Kd + k0];
    v16bf a0 = load_frag2(ar0 + (hi ? 8 : 0), ar0 + 16 + (hi ? 8 : 0));
    v16bf a1 = load_frag2(ar1 + (hi ? 8 : 0), ar1 + 16 + (hi ? 8 : 0));

    if (t0ok) {
      const u16* brow = Wt + (size_t)(nb + ln) * Kd + k0;
      v16bf b0 = load_frag2(brow + (hi ? 8 : 0), brow + 16 + (hi ? 8 : 0));
      acc00 = wmma_bf16(a0, b0, acc00);
      acc10 = wmma_bf16(a1, b0, acc10);
    }
    if (t1ok) {
      const u16* brow = Wt + (size_t)(nb + 16 + ln) * Kd + k0;
      v16bf b1 = load_frag2(brow + (hi ? 8 : 0), brow + 16 + (hi ? 8 : 0));
      acc01 = wmma_bf16(a0, b1, acc01);
      acc11 = wmma_bf16(a1, b1, acc11);
    }
    if (k0 + 32 < Kd && t0ok) {  // global_prefetch_b8 for next B tile
      __builtin_prefetch(Wt + (size_t)(nb + ln) * Kd + k0 + 32, 0, 3);
      if (t1ok) __builtin_prefetch(Wt + (size_t)(nb + 16 + ln) * Kd + k0 + 32, 0, 3);
    }
  }

  for (int im = 0; im < 2; ++im) {
    for (int t = 0; t < 2; ++t) {
      if (t == 0 && !t0ok) continue;
      if (t == 1 && !t1ok) continue;
      v8f acc = (im == 0) ? (t ? acc01 : acc00) : (t ? acc11 : acc10);
      int n = nb + t * 16 + ln;
      float bv = bias[n];
#pragma unroll
      for (int r = 0; r < 8; ++r) {
        int m = m0 + waveM * 32 + im * 16 + r + hi * 8;
        float v = acc[r] + bv;
        if (EPI == 0) {
          ((u16*)out)[(size_t)m * Ncols + n] = f2bf(v);
        } else if (EPI == 1) {
          float gv = 0.5f * v * (1.0f + erff(v * 0.70710678118654752f));
          ((u16*)out)[(size_t)m * Ncols + n] = f2bf(gv);
        } else if (EPI == 2) {
          // m is windowed row (bw, n_in_win); reverse windows + un-shift.
          int bw = m / NWIN, nwp = m % NWIN;
          int bb = bw / (NZW * NHW * NWW); int r1 = bw % (NZW * NHW * NWW);
          int wz = r1 / (NHW * NWW);       int r2 = r1 % (NHW * NWW);
          int wh = r2 / NWW, ww = r2 % NWW;
          int lz = nwp / (WH * WWIN), lh = (nwp / WWIN) % WH, lw = nwp % WWIN;
          int z = wz * WZ + lz + SZ;   if (z >= ZD)  z -= ZD;
          int h = wh * WH + lh + SH;   if (h >= HD_) h -= HD_;
          int w = ww * WWIN + lw + SW; if (w >= WD)  w -= WD;
          size_t pos = (((size_t)bb * ZD + z) * HD_ + h) * WD + w;
          float sc = resid[((size_t)bb * CC + n) * SPAT + (size_t)z * HD_ * WD +
                           h * WD + w];                       // shortcut from x
          ((float*)out)[pos * CC + n] = v + sc;
        } else {  // EPI == 3 : fc2 + residual + CF transpose
          float ov = v + resid[(size_t)m * CC + n];
          int bb = m / SPAT, sp = m % SPAT;
          ((float*)out)[((size_t)bb * CC + n) * SPAT + sp] = ov;
        }
      }
    }
  }
}

// ---------------------------------------------------------------------------
// Attention: one block per (window, head, 32-row tile). 256 thr = 8 waves.
//   S = scale*Q@K^T + rel-pos bias + shift mask  (40 WMMA tiles, K=32)
//   softmax over 320 keys in LDS
//   O = P@V (V staged transposed in LDS), 4 waves x 10 WMMAs
// ---------------------------------------------------------------------------
__global__ __launch_bounds__(256) void attn_kernel(
    const u16* __restrict__ qkv, const float* __restrict__ rpb,
    u16* __restrict__ aout) {
  __shared__ float Sf[32][NWIN];
  __shared__ u16   Pt[32][NWIN];
  __shared__ u16   Vt[HDIM][NWIN];
  __shared__ float red[32][8];
  __shared__ float rowstat[32];

  int blk = blockIdx.x;
  int mt = blk % 10; int t = blk / 10;
  int head = t % NHEAD; int bw = t / NHEAD;

  const int tid = threadIdx.x;
  const int wave = tid >> 5, lane = tid & 31, hi = lane >> 4, ln = lane & 15;

  const u16* qbase = qkv + (size_t)bw * NWIN * C3 + head * HDIM;
  const u16* kbase = qbase + CC;
  const u16* vbase = qbase + 2 * CC;

  // stage V transposed: Vt[d][key]
  for (int i = tid; i < NWIN * HDIM; i += 256) {
    int key = i >> 5, d = i & 31;
    Vt[d][key] = vbase[(size_t)key * C3 + d];
  }

  // window position for the shift mask
  int r1 = bw % (NZW * NHW * NWW);
  int wz = r1 / (NHW * NWW); int r2 = r1 % (NHW * NWW);
  int wh = r2 / NWW, ww = r2 % NWW;
  const float scale = 0.17677669529663687f;  // 1/sqrt(32)

  // ---- phase 1: logits -------------------------------------------------
  for (int tt = wave; tt < 40; tt += 8) {
    int mrow = tt & 1, ncol = tt >> 1;
    const u16* arow = qbase + (size_t)(mt * 32 + mrow * 16 + ln) * C3;
    v16bf afrag = load_frag2(arow + (hi ? 8 : 0), arow + 16 + (hi ? 8 : 0));
    const u16* brow = kbase + (size_t)(ncol * 16 + ln) * C3;
    v16bf bfrag = load_frag2(brow + (hi ? 8 : 0), brow + 16 + (hi ? 8 : 0));
    v8f z = {};
    v8f s = wmma_bf16(afrag, bfrag, z);

    int kn = ncol * 16 + ln;
    int kz = kn >> 6, kh = (kn >> 3) & 7, kw = kn & 7;
    int kzg = wz * WZ + kz, khg = wh * WH + kh, kwg = ww * WWIN + kw;
    int cntk = (kzg < (ZD - WZ) ? 0 : (kzg < (ZD - SZ) ? 1 : 2)) * 9 +
               (khg < (HD_ - WH) ? 0 : (khg < (HD_ - SH) ? 1 : 2)) * 3 +
               (kwg < (WD - WWIN) ? 0 : (kwg < (WD - SW) ? 1 : 2));
#pragma unroll
    for (int r = 0; r < 8; ++r) {
      int ml = mrow * 16 + r + hi * 8;          // row inside 32-row tile
      int qn = mt * 32 + ml;
      int qz = qn >> 6, qh = (qn >> 3) & 7, qw = qn & 7;
      int idx = (qz - kz + WZ - 1) * ((2 * WH - 1) * (2 * WWIN - 1)) +
                (qh - kh + WH - 1) * (2 * WWIN - 1) + (qw - kw + WWIN - 1);
      float bia = rpb[idx * NHEAD + head];
      int qzg = wz * WZ + qz, qhg = wh * WH + qh, qwg = ww * WWIN + qw;
      int cntq = (qzg < (ZD - WZ) ? 0 : (qzg < (ZD - SZ) ? 1 : 2)) * 9 +
                 (qhg < (HD_ - WH) ? 0 : (qhg < (HD_ - SH) ? 1 : 2)) * 3 +
                 (qwg < (WD - WWIN) ? 0 : (qwg < (WD - SW) ? 1 : 2));
      Sf[ml][kn] = s[r] * scale + bia + ((cntq != cntk) ? -100.0f : 0.0f);
    }
  }
  __syncthreads();

  // ---- phase 2: softmax -------------------------------------------------
  {
    int row = tid >> 3, sub = tid & 7;
    float lmax = -1e30f;
    for (int c = sub; c < NWIN; c += 8) { float v = Sf[row][c]; lmax = v > lmax ? v : lmax; }
    red[row][sub] = lmax;
    __syncthreads();
    if (sub == 0) {
      float m = red[row][0];
      for (int i = 1; i < 8; ++i) m = red[row][i] > m ? red[row][i] : m;
      rowstat[row] = m;
    }
    __syncthreads();
    float m = rowstat[row];
    float lsum = 0.f;
    for (int c = sub; c < NWIN; c += 8) { float e = __expf(Sf[row][c] - m); Sf[row][c] = e; lsum += e; }
    red[row][sub] = lsum;
    __syncthreads();
    if (sub == 0) {
      float s2 = 0.f;
      for (int i = 0; i < 8; ++i) s2 += red[row][i];
      rowstat[row] = 1.0f / s2;
    }
    __syncthreads();
    float inv = rowstat[row];
    for (int c = sub; c < NWIN; c += 8) Pt[row][c] = f2bf(Sf[row][c] * inv);
  }
  __syncthreads();

  // ---- phase 3: O = P @ V ----------------------------------------------
  if (wave < 4) {
    int mrow = wave & 1, dt = wave >> 1;
    v8f acc = {};
    for (int kt = 0; kt < 10; ++kt) {
      const u16* arow = &Pt[mrow * 16 + ln][kt * 32];
      v16bf afrag = load_frag2(arow + (hi ? 8 : 0), arow + 16 + (hi ? 8 : 0));
      const u16* brow = &Vt[dt * 16 + ln][kt * 32];
      v16bf bfrag = load_frag2(brow + (hi ? 8 : 0), brow + 16 + (hi ? 8 : 0));
      acc = wmma_bf16(afrag, bfrag, acc);
    }
#pragma unroll
    for (int r = 0; r < 8; ++r) {
      int qn = mt * 32 + mrow * 16 + r + hi * 8;
      aout[((size_t)bw * NWIN + qn) * CC + head * HDIM + dt * 16 + ln] = f2bf(acc[r]);
    }
  }
}

// ---------------------------------------------------------------------------
// LayerNorm2: one wave per token (channels-last f32, fully coalesced) -> bf16
// ---------------------------------------------------------------------------
__global__ void ln2_kernel(const float* __restrict__ hbuf,
                           const float* __restrict__ g,
                           const float* __restrict__ b,
                           u16* __restrict__ outb) {
  int row = (blockIdx.x * blockDim.x + threadIdx.x) >> 5;
  int lane = threadIdx.x & 31;
  if (row >= MTOK) return;
  const float* rp = hbuf + (size_t)row * CC;
  float s = 0.f, ss = 0.f;
  for (int c = lane; c < CC; c += 32) { float v = rp[c]; s += v; ss += v * v; }
  for (int off = 16; off > 0; off >>= 1) {
    s += __shfl_xor(s, off, 32);
    ss += __shfl_xor(ss, off, 32);
  }
  float mu = s * (1.0f / CC);
  float var = ss * (1.0f / CC) - mu * mu;
  float rs = rsqrtf(var + 1e-5f);
  u16* op = outb + (size_t)row * CC;
  for (int c = lane; c < CC; c += 32) op[c] = f2bf((rp[c] - mu) * rs * g[c] + b[c]);
}

// ---------------------------------------------------------------------------
// Launcher
// ---------------------------------------------------------------------------
extern "C" void kernel_launch(void* const* d_in, const int* in_sizes, int n_in,
                              void* d_out, int out_size, void* d_ws, size_t ws_size,
                              hipStream_t stream) {
  (void)in_sizes; (void)n_in; (void)out_size; (void)ws_size;

  const float* x       = (const float*)d_in[0];
  const float* n1g     = (const float*)d_in[1];
  const float* n1b     = (const float*)d_in[2];
  const float* qkv_w   = (const float*)d_in[3];
  const float* qkv_b   = (const float*)d_in[4];
  const float* proj_w  = (const float*)d_in[5];
  const float* proj_b  = (const float*)d_in[6];
  const float* rpb     = (const float*)d_in[7];
  const float* n2g     = (const float*)d_in[8];
  const float* n2b     = (const float*)d_in[9];
  const float* fc1_w   = (const float*)d_in[10];
  const float* fc1_b   = (const float*)d_in[11];
  const float* fc2_w   = (const float*)d_in[12];
  const float* fc2_b   = (const float*)d_in[13];

  char* ws = (char*)d_ws;
  size_t off = 0;
  auto take = [&](size_t bytes) {
    size_t o = off;
    off = (off + bytes + 255) & ~(size_t)255;
    return o;
  };
  u16* wq = (u16*)(ws + take((size_t)C3 * CC * 2));       // qkv_w bf16
  u16* wp = (u16*)(ws + take((size_t)CC * CC * 2));       // proj_w bf16
  u16* w1 = (u16*)(ws + take((size_t)HID * CC * 2));      // fc1_w bf16
  u16* w2 = (u16*)(ws + take((size_t)CC * HID * 2));      // fc2_w bf16
  u16* xwin   = (u16*)(ws + take((size_t)NWTOT * NWIN * CC * 2)); // also h2ln
  u16* bigbuf = (u16*)(ws + take((size_t)MTOK * HID * 2));        // qkv out & fc1 out
  u16* aout   = (u16*)(ws + take((size_t)NWTOT * NWIN * CC * 2));
  float* hbuf = (float*)(ws + take((size_t)MTOK * CC * 4));
  u16* h2ln   = xwin;      // alias: xwin dead after attention input consumed
  u16* qkvbuf = bigbuf;    // alias
  u16* fc1out = bigbuf;    // alias (written after qkvbuf is dead)

  // 1) weights -> bf16
  cvt_f32_bf16<<<(C3 * CC + 255) / 256, 256, 0, stream>>>(qkv_w, wq, C3 * CC);
  cvt_f32_bf16<<<(CC * CC + 255) / 256, 256, 0, stream>>>(proj_w, wp, CC * CC);
  cvt_f32_bf16<<<(HID * CC + 255) / 256, 256, 0, stream>>>(fc1_w, w1, HID * CC);
  cvt_f32_bf16<<<(CC * HID + 255) / 256, 256, 0, stream>>>(fc2_w, w2, CC * HID);

  // 2) LN1 + shift + window partition
  ln1_window_kernel<<<(MTOK + 255) / 256, 256, 0, stream>>>(x, n1g, n1b, xwin);

  // dynamic LDS for the TDM-staged A tile: 64 rows x K bf16
  const size_t lds_k192 = 64 * 192 * sizeof(u16);   // 24 KB
  const size_t lds_k768 = 64 * 768 * sizeof(u16);   // 96 KB

  // 3) QKV GEMM: 46080 x 576 x 192
  gemm_wmma<0><<<dim3(MTOK / 64, 5), 256, lds_k192, stream>>>(
      xwin, wq, qkv_b, C3, CC, qkvbuf, nullptr);

  // 4) windowed attention (144 windows x 6 heads x 10 row-tiles)
  attn_kernel<<<NWTOT * NHEAD * 10, 256, 0, stream>>>(qkvbuf, rpb, aout);

  // 5) proj GEMM + window-reverse + residual -> hbuf (f32, channels-last)
  gemm_wmma<2><<<dim3(MTOK / 64, 2), 256, lds_k192, stream>>>(
      aout, wp, proj_b, CC, CC, hbuf, x);

  // 6) LN2 -> bf16
  ln2_kernel<<<(MTOK * 32) / 256, 256, 0, stream>>>(hbuf, n2g, n2b, h2ln);

  // 7) FC1 + GELU: 46080 x 768 x 192
  gemm_wmma<1><<<dim3(MTOK / 64, 6), 256, lds_k192, stream>>>(
      h2ln, w1, fc1_b, HID, CC, fc1out, nullptr);

  // 8) FC2 + residual + transpose to (B,C,Z,H,W): 46080 x 192 x 768
  gemm_wmma<3><<<dim3(MTOK / 64, 2), 256, lds_k768, stream>>>(
      fc1out, w2, fc2_b, CC, HID, (float*)d_out, hbuf);
}
